// VOLoss_44813688766950
// MI455X (gfx1250) — compile-verified, matrix operand order checked by reading, not run on hardware
//
#include <hip/hip_runtime.h>
#include <hip/hip_bf16.h>
#include <math.h>

#define BATCH 8
#define NPTS  1024
#define DDIM  256
#define MTOT  (NPTS*NPTS)
#define FEPS  1e-8f
#define NBINS 4096
#define KRANK 838860   /* floor(0.8*(MTOT-1)); fractional part is exactly 0 */

typedef __attribute__((ext_vector_type(16))) _Float16 v16h;
typedef __attribute__((ext_vector_type(8)))  _Float16 v8h;
typedef __attribute__((ext_vector_type(8)))  float    v8f;

// ---------------- workspace layout (bytes) ----------------
constexpr size_t WS_DESC1  = 0;                                       // B*N*D f16
constexpr size_t WS_DESC2  = WS_DESC1 + (size_t)BATCH*NPTS*DDIM*2;    // B*N*D f16
constexpr size_t WS_ROWMAX = WS_DESC2 + (size_t)BATCH*NPTS*DDIM*2;    // B*N f32
constexpr size_t WS_P2T    = WS_ROWMAX + (size_t)BATCH*NPTS*4;        // B*N*2 f32
constexpr size_t WS_FX1    = WS_P2T    + (size_t)BATCH*NPTS*8;        // B*N*3 f32
constexpr size_t WS_D1S    = WS_FX1    + (size_t)BATCH*NPTS*12;       // B*N f32
constexpr size_t WS_D2S    = WS_D1S    + (size_t)BATCH*NPTS*4;        // B*N f32
constexpr size_t WS_HIST   = WS_D2S    + (size_t)BATCH*NPTS*4;        // B*4096 u32
constexpr size_t WS_QINFO  = WS_HIST   + (size_t)BATCH*NBINS*4;       // B*4 f32
constexpr size_t WS_THR    = WS_QINFO  + (size_t)BATCH*16;            // B f32
constexpr size_t WS_MM     = WS_THR    + (size_t)BATCH*4;             // B*2 u32 keys
constexpr size_t WS_F      = WS_MM     + (size_t)BATCH*8;             // B*9 f32
constexpr size_t WS_ACC    = WS_F      + (size_t)BATCH*36;            // B*9 f32
constexpr size_t WS_POSE   = WS_ACC    + (size_t)BATCH*36;            // 1 f32

__device__ __forceinline__ unsigned f2key(float f) {
  unsigned u = __float_as_uint(f);
  return (u & 0x80000000u) ? ~u : (u | 0x80000000u);
}
__device__ __forceinline__ float key2f(unsigned k) {
  unsigned u = (k & 0x80000000u) ? (k ^ 0x80000000u) : ~k;
  return __uint_as_float(u);
}

// ---------------- init: zero histograms/accumulators, init min/max keys ----
__global__ void init_kernel(unsigned* __restrict__ ghist, float* __restrict__ acc,
                            unsigned* __restrict__ mm, float* __restrict__ pose) {
  int tid = blockIdx.x * blockDim.x + threadIdx.x;
  if (tid < BATCH * NBINS) ghist[tid] = 0u;
  if (tid < BATCH * 9)     acc[tid] = 0.f;
  if (tid < BATCH) { mm[tid*2] = 0xFFFFFFFFu; mm[tid*2+1] = 0u; }
  if (tid == 0) pose[0] = 0.f;
}

// ---------------- f32 -> f16 descriptor conversion ----------------
__global__ void convert_kernel(const float* __restrict__ d1, const float* __restrict__ d2,
                               _Float16* __restrict__ o1, _Float16* __restrict__ o2) {
  int tid = blockIdx.x * blockDim.x + threadIdx.x;
  if (tid < BATCH * NPTS * DDIM) {
    o1[tid] = (_Float16)d1[tid];
    o2[tid] = (_Float16)d2[tid];
  }
}

// ---------------- per-row max of matches + global per-batch min/max --------
__global__ void rowmax_kernel(const float* __restrict__ matches,
                              float* __restrict__ rowmax, unsigned* __restrict__ mm) {
  __shared__ float smax[256], smin[256];
  const int row = blockIdx.x;            // b*N + n
  const int b = row >> 10;
  const size_t base = (size_t)row * NPTS;
  float mx = -3.4e38f, mn = 3.4e38f;
  for (int j = threadIdx.x; j < NPTS; j += 256) {
    float v = matches[base + j];
    mx = fmaxf(mx, v); mn = fminf(mn, v);
  }
  smax[threadIdx.x] = mx; smin[threadIdx.x] = mn;
  __syncthreads();
  for (int s = 128; s > 0; s >>= 1) {
    if (threadIdx.x < s) {
      smax[threadIdx.x] = fmaxf(smax[threadIdx.x], smax[threadIdx.x + s]);
      smin[threadIdx.x] = fminf(smin[threadIdx.x], smin[threadIdx.x + s]);
    }
    __syncthreads();
  }
  if (threadIdx.x == 0) {
    rowmax[row] = smax[0];
    atomicMax(&mm[b*2 + 1], f2key(smax[0]));
    atomicMin(&mm[b*2 + 0], f2key(smin[0]));
  }
}

// ---------------- quantile: level-1 histogram ----------------
__global__ void hist1_kernel(const float* __restrict__ matches,
                             const unsigned* __restrict__ mm, unsigned* __restrict__ ghist) {
  __shared__ unsigned sh[NBINS];
  const int b = blockIdx.x >> 5;
  const int chunk = blockIdx.x & 31;
  for (int t = threadIdx.x; t < NBINS; t += blockDim.x) sh[t] = 0u;
  __syncthreads();
  const float mn = key2f(mm[b*2 + 0]);
  const float mx = key2f(mm[b*2 + 1]);
  const float scale = (float)NBINS / fmaxf(mx - mn, 1e-30f);
  const size_t base = (size_t)b * MTOT + (size_t)chunk * (MTOT / 32);
  for (int t = threadIdx.x; t < MTOT / 32; t += blockDim.x) {
    float v = matches[base + t];
    int bin = (int)((v - mn) * scale);
    bin = bin < 0 ? 0 : (bin > NBINS - 1 ? NBINS - 1 : bin);
    atomicAdd(&sh[bin], 1u);
  }
  __syncthreads();
  for (int t = threadIdx.x; t < NBINS; t += blockDim.x)
    if (sh[t]) atomicAdd(&ghist[b*NBINS + t], sh[t]);
}

// ---------------- quantile: find level-1 bin containing rank, zero hist ---
__global__ void scan1_kernel(unsigned* __restrict__ ghist,
                             const unsigned* __restrict__ mm, float* __restrict__ qinfo) {
  const int b = blockIdx.x;
  if (threadIdx.x == 0) {
    float mn = key2f(mm[b*2 + 0]);
    float mx = key2f(mm[b*2 + 1]);
    float w  = fmaxf(mx - mn, 1e-30f) / (float)NBINS;
    long cum = 0; int bin = NBINS - 1; long rem = 0;
    for (int t = 0; t < NBINS; ++t) {
      long c = (long)ghist[b*NBINS + t];
      if (cum + c > (long)KRANK) { bin = t; rem = (long)KRANK - cum; break; }
      cum += c;
    }
    qinfo[b*4 + 0] = mn + bin * w;
    qinfo[b*4 + 1] = w;
    qinfo[b*4 + 2] = (float)rem;
    qinfo[b*4 + 3] = (float)bin;
  }
  __syncthreads();
  for (int t = threadIdx.x; t < NBINS; t += blockDim.x) ghist[b*NBINS + t] = 0u;
}

// ---------------- quantile: level-2 histogram inside selected bin ---------
__global__ void hist2_kernel(const float* __restrict__ matches,
                             const unsigned* __restrict__ mm,
                             const float* __restrict__ qinfo, unsigned* __restrict__ ghist) {
  __shared__ unsigned sh[NBINS];
  const int b = blockIdx.x >> 5;
  const int chunk = blockIdx.x & 31;
  for (int t = threadIdx.x; t < NBINS; t += blockDim.x) sh[t] = 0u;
  __syncthreads();
  const float mn = key2f(mm[b*2 + 0]);
  const float mx = key2f(mm[b*2 + 1]);
  const float scale1 = (float)NBINS / fmaxf(mx - mn, 1e-30f);
  const int   target = (int)qinfo[b*4 + 3];
  const float qlo    = qinfo[b*4 + 0];
  const float qw     = qinfo[b*4 + 1];
  const float scale2 = (float)NBINS / fmaxf(qw, 1e-37f);
  const size_t base = (size_t)b * MTOT + (size_t)chunk * (MTOT / 32);
  for (int t = threadIdx.x; t < MTOT / 32; t += blockDim.x) {
    float v = matches[base + t];
    int bin = (int)((v - mn) * scale1);
    bin = bin < 0 ? 0 : (bin > NBINS - 1 ? NBINS - 1 : bin);
    if (bin == target) {
      int sub = (int)((v - qlo) * scale2);
      sub = sub < 0 ? 0 : (sub > NBINS - 1 ? NBINS - 1 : sub);
      atomicAdd(&sh[sub], 1u);
    }
  }
  __syncthreads();
  for (int t = threadIdx.x; t < NBINS; t += blockDim.x)
    if (sh[t]) atomicAdd(&ghist[b*NBINS + t], sh[t]);
}

// ---------------- quantile: pick sub-bin -> threshold ----------------
__global__ void scan2_kernel(const unsigned* __restrict__ ghist,
                             const float* __restrict__ qinfo, float* __restrict__ thr) {
  const int b = blockIdx.x;
  if (threadIdx.x != 0) return;
  const float qlo = qinfo[b*4 + 0];
  const float qw  = qinfo[b*4 + 1];
  const long rem  = (long)qinfo[b*4 + 2];
  const float w2  = qw / (float)NBINS;
  long cum = 0; int sub = NBINS - 1;
  for (int t = 0; t < NBINS; ++t) {
    long c = (long)ghist[b*NBINS + t];
    if (cum + c > rem) { sub = t; break; }
    cum += c;
  }
  thr[b] = qlo + sub * w2;
}

// ---------------- pose losses + fundamental matrices ----------------
__global__ __launch_bounds__(32) void poseF_kernel(
    const float* __restrict__ Rp, const float* __restrict__ tp,
    const float* __restrict__ Rg, const float* __restrict__ tg,
    const float* __restrict__ Kin, float* __restrict__ Fm, float* __restrict__ pose) {
  __shared__ float sl[BATCH], so[BATCH], st[BATCH];
  const int b = threadIdx.x;
  if (b < BATCH) {
    const float* A = Rp + b*9;
    const float* G = Rg + b*9;
    float trace = 0.f;
    for (int k = 0; k < 9; ++k) trace += A[k] * G[k];
    float geo = fminf(fmaxf((trace - 1.f) * 0.5f, -0.9999f), 0.9999f);
    sl[b] = 1.f - geo;
    float s = 0.f;
    for (int i = 0; i < 3; ++i)
      for (int j = 0; j < 3; ++j) {
        float v = A[i*3+0]*A[j*3+0] + A[i*3+1]*A[j*3+1] + A[i*3+2]*A[j*3+2];
        if (i == j) v -= 1.f;
        s += v * v;
      }
    so[b] = sqrtf(s);
    const float* t1 = tp + b*3; const float* t2 = tg + b*3;
    float n1 = fmaxf(sqrtf(t1[0]*t1[0] + t1[1]*t1[1] + t1[2]*t1[2]), 1e-12f);
    float n2 = fmaxf(sqrtf(t2[0]*t2[0] + t2[1]*t2[1] + t2[2]*t2[2]), 1e-12f);
    float cosv = (t1[0]*t2[0] + t1[1]*t2[1] + t1[2]*t2[2]) / (n1 * n2);
    cosv = fminf(fmaxf(cosv, -1.f + 1e-7f), 1.f - 1e-7f);
    st[b] = acosf(cosv);
    // fundamental matrix F = Kinv^T (skew(tn) Rg) Kinv
    float tn0 = t2[0]/n2, tn1 = t2[1]/n2, tn2 = t2[2]/n2;
    float S[9] = {0.f, -tn2, tn1, tn2, 0.f, -tn0, -tn1, tn0, 0.f};
    float E[9];
    for (int i = 0; i < 3; ++i)
      for (int j = 0; j < 3; ++j)
        E[i*3+j] = S[i*3+0]*G[0*3+j] + S[i*3+1]*G[1*3+j] + S[i*3+2]*G[2*3+j];
    const float* K = Kin + b*9;
    float a = K[0], bb = K[1], c = K[2], d = K[3], e = K[4], f = K[5],
          g = K[6], h = K[7], ii = K[8];
    float det = a*(e*ii - f*h) - bb*(d*ii - f*g) + c*(d*h - e*g);
    float id = 1.f / det;
    float KI[9] = {
      (e*ii - f*h)*id, (c*h - bb*ii)*id, (bb*f - c*e)*id,
      (f*g - d*ii)*id, (a*ii - c*g)*id,  (c*d - a*f)*id,
      (d*h - e*g)*id,  (bb*g - a*h)*id,  (a*e - bb*d)*id };
    float T[9];
    for (int i = 0; i < 3; ++i)
      for (int j = 0; j < 3; ++j)
        T[i*3+j] = E[i*3+0]*KI[0*3+j] + E[i*3+1]*KI[1*3+j] + E[i*3+2]*KI[2*3+j];
    for (int i = 0; i < 3; ++i)
      for (int j = 0; j < 3; ++j)
        Fm[b*9 + i*3 + j] = KI[0*3+i]*T[0*3+j] + KI[1*3+i]*T[1*3+j] + KI[2*3+i]*T[2*3+j];
  }
  __syncthreads();
  if (threadIdx.x == 0) {
    float l = 0.f, o = 0.f, t = 0.f;
    for (int k = 0; k < BATCH; ++k) { l += sl[k]; o += so[k]; t += st[k]; }
    pose[0] = l / BATCH + 0.5f * (o / BATCH) + t / BATCH;
  }
}

// ---------------- per-keypoint precompute (p2t, F*p1h, denominators) ------
__global__ void precompute_kernel(const float* __restrict__ kp1, const float* __restrict__ kp2,
                                  const float* __restrict__ Rgt, const float* __restrict__ Fm,
                                  float* __restrict__ p2t, float* __restrict__ fx1,
                                  float* __restrict__ d1s, float* __restrict__ d2s) {
  int tid = blockIdx.x * blockDim.x + threadIdx.x;
  if (tid >= BATCH * NPTS) return;
  const int b = tid >> 10;
  float x1 = kp1[tid*2], y1 = kp1[tid*2 + 1];
  float x2 = kp2[tid*2], y2 = kp2[tid*2 + 1];
  const float* R = Rgt + b*9;
  p2t[tid*2 + 0] = R[0]*x1 + R[1]*y1 + R[2];
  p2t[tid*2 + 1] = R[3]*x1 + R[4]*y1 + R[5];
  const float* F = Fm + b*9;
  float f0 = F[0]*x1 + F[1]*y1 + F[2];
  float f1 = F[3]*x1 + F[4]*y1 + F[5];
  float f2 = F[6]*x1 + F[7]*y1 + F[8];
  fx1[tid*3 + 0] = f0; fx1[tid*3 + 1] = f1; fx1[tid*3 + 2] = f2;
  d1s[tid] = f0*f0 + f1*f1;
  float g0 = F[0]*x2 + F[3]*y2 + F[6];
  float g1 = F[1]*x2 + F[4]*y2 + F[7];
  d2s[tid] = g0*g0 + g1*g1;
}

// ---------------- fused correspondence + epipolar streaming pass ----------
__global__ void corr_epi_kernel(const float* __restrict__ matches, const float* __restrict__ kp2,
                                const float* __restrict__ p2t, const float* __restrict__ fx1,
                                const float* __restrict__ d1s, const float* __restrict__ d2s,
                                const float* __restrict__ thr, float* __restrict__ acc) {
  __shared__ float sacc[6];
  const int row = blockIdx.x;     // b*N + i
  const int b = row >> 10;
  if (threadIdx.x < 6) sacc[threadIdx.x] = 0.f;
  __syncthreads();
  const float px = p2t[row*2], py = p2t[row*2 + 1];
  const float f0 = fx1[row*3], f1 = fx1[row*3 + 1], f2 = fx1[row*3 + 2];
  const float d1 = d1s[row];
  const float tb = thr[b];
  const size_t base = (size_t)row * NPTS;
  const size_t kb = (size_t)b * NPTS;
  float cw = 0.f, cm = 0.f, cc = 0.f, ew = 0.f, ec = 0.f, en = 0.f;
  for (int j = threadIdx.x; j < NPTS; j += 256) {
    float m  = matches[base + j];
    float kx = kp2[(kb + j)*2], ky = kp2[(kb + j)*2 + 1];
    float dx = kx - px, dy = ky - py;
    cw += m * sqrtf(dx*dx + dy*dy);
    cm += m;
    cc += (m > 0.01f) ? 1.f : 0.f;
    float num = f0*kx + f1*ky + f2;
    num *= num;
    float samp = num / (d1 + d2s[kb + j] + FEPS);
    if (m > tb) { ew += samp * m; ec += m; en += 1.f; }
  }
  atomicAdd(&sacc[0], cw); atomicAdd(&sacc[1], cm); atomicAdd(&sacc[2], cc);
  atomicAdd(&sacc[3], ew); atomicAdd(&sacc[4], ec); atomicAdd(&sacc[5], en);
  __syncthreads();
  if (threadIdx.x < 6) atomicAdd(&acc[b*9 + threadIdx.x], sacc[threadIdx.x]);
}

// ---------------- contrastive loss: WMMA f16 GEMM + fused softmax sums ----
__global__ __launch_bounds__(32) void contrastive_kernel(
    const float* __restrict__ matches, const _Float16* __restrict__ d1h,
    const _Float16* __restrict__ d2h, const float* __restrict__ rowmax,
    float* __restrict__ acc) {
  const int lane = threadIdx.x;
  const int b  = blockIdx.x >> 6;
  const int it = blockIdx.x & 63;
  const int i0 = it * 16;
  const int hf = lane >> 4;
  const int lm = lane & 15;
  const float inv_temp = 1.0f / 0.07f;

  // A fragments: 16 rows of desc1, K=256 split into 8 chunks of 32
  const _Float16* arow = d1h + ((size_t)(b*NPTS + i0 + lm)) * DDIM + hf*8;
  v16h afrag[8];
#pragma unroll
  for (int kc = 0; kc < 8; ++kc) {
    const _Float16* p = arow + kc*32;
    v8h lo = *reinterpret_cast<const v8h*>(p);
    v8h hi = *reinterpret_cast<const v8h*>(p + 16);
    afrag[kc] = __builtin_shufflevector(lo, hi, 0,1,2,3,4,5,6,7,8,9,10,11,12,13,14,15);
  }

  float rm[8];
#pragma unroll
  for (int r = 0; r < 8; ++r)
    rm[r] = 0.5f * rowmax[b*NPTS + i0 + hf*8 + r];

  float accAll[8], accPos[8], accCnt[8];
#pragma unroll
  for (int r = 0; r < 8; ++r) { accAll[r] = 0.f; accPos[r] = 0.f; accCnt[r] = 0.f; }

  for (int jt = 0; jt < 64; ++jt) {
    const int j0 = jt * 16;
    const _Float16* brow = d2h + ((size_t)(b*NPTS + j0 + lm)) * DDIM + hf*8;
    v8f c = {};
#pragma unroll
    for (int kc = 0; kc < 8; ++kc) {
      const _Float16* p = brow + kc*32;
      v8h lo = *reinterpret_cast<const v8h*>(p);
      v8h hi = *reinterpret_cast<const v8h*>(p + 16);
      v16h bfrag = __builtin_shufflevector(lo, hi, 0,1,2,3,4,5,6,7,8,9,10,11,12,13,14,15);
      c = __builtin_amdgcn_wmma_f32_16x16x32_f16(false, afrag[kc], false, bfrag,
                                                 (short)0, c, false, false);
    }
    const int j = j0 + lm;
#pragma unroll
    for (int r = 0; r < 8; ++r) {
      const int i = i0 + hf*8 + r;
      float sim = c[r] * inv_temp;
      float m = matches[((size_t)(b*NPTS) + i) * NPTS + j];
      float e = __expf(sim);
      accAll[r] += e;
      if (m > rm[r]) { accPos[r] += e; accCnt[r] += 1.f; }
    }
  }
  // reduce over the 16 lanes of each half-wave (N dimension of the tile)
#pragma unroll
  for (int r = 0; r < 8; ++r) {
#pragma unroll
    for (int mks = 8; mks >= 1; mks >>= 1) {
      accAll[r] += __shfl_xor(accAll[r], mks);
      accPos[r] += __shfl_xor(accPos[r], mks);
      accCnt[r] += __shfl_xor(accCnt[r], mks);
    }
  }
#pragma unroll
  for (int r = 0; r < 8; ++r) {
    if (lm == r) {
      float valid = (accCnt[r] > 0.f) ? 1.f : 0.f;
      float lpp = -logf((accPos[r] + FEPS) / (accAll[r] + FEPS));
      atomicAdd(&acc[b*9 + 6], lpp * valid);
      atomicAdd(&acc[b*9 + 7], valid);
      atomicAdd(&acc[b*9 + 8], accCnt[r]);
    }
  }
}

// ---------------- final reduction to scalar ----------------
__global__ void finalize_kernel(const float* __restrict__ acc,
                                const float* __restrict__ pose, float* __restrict__ out) {
  if (threadIdx.x == 0 && blockIdx.x == 0) {
    float cs = 0.f, ch = 0.f, es = 0.f, eh = 0.f, ss = 0.f, sh = 0.f;
    for (int b = 0; b < BATCH; ++b) {
      const float* a = acc + b*9;
      float h1 = (a[2] >= 2.5f) ? 1.f : 0.f;    // correspondence has-mask
      cs += h1 * (a[0] / (a[1] + FEPS)); ch += h1;
      float h2 = (a[5] >= 2.5f) ? 1.f : 0.f;    // epipolar has-mask
      es += h2 * (a[3] / (a[4] + FEPS)); eh += h2;
      float h3 = (a[8] > 0.5f) ? 1.f : 0.f;     // contrastive has-mask
      ss += h3 * (a[6] / (a[7] + FEPS)); sh += h3;
    }
    float corr = cs / (ch + FEPS);
    float epi  = es / (eh + FEPS);
    float con  = ss / (sh + FEPS);
    out[0] = pose[0] + 0.05f * corr + 0.02f * epi + 0.01f * con;
  }
}

extern "C" void kernel_launch(void* const* d_in, const int* in_sizes, int n_in,
                              void* d_out, int out_size, void* d_ws, size_t ws_size,
                              hipStream_t stream) {
  const float* R_pred  = (const float*)d_in[0];
  const float* t_pred  = (const float*)d_in[1];
  const float* kp1     = (const float*)d_in[2];
  const float* kp2     = (const float*)d_in[3];
  const float* matches = (const float*)d_in[4];
  const float* desc1   = (const float*)d_in[5];
  const float* desc2   = (const float*)d_in[6];
  const float* R_gt    = (const float*)d_in[7];
  const float* t_gt    = (const float*)d_in[8];
  const float* Kmat    = (const float*)d_in[9];

  char* ws = (char*)d_ws;
  _Float16* d1h   = (_Float16*)(ws + WS_DESC1);
  _Float16* d2h   = (_Float16*)(ws + WS_DESC2);
  float*    rmax  = (float*)(ws + WS_ROWMAX);
  float*    p2t   = (float*)(ws + WS_P2T);
  float*    fx1   = (float*)(ws + WS_FX1);
  float*    d1s   = (float*)(ws + WS_D1S);
  float*    d2s   = (float*)(ws + WS_D2S);
  unsigned* ghist = (unsigned*)(ws + WS_HIST);
  float*    qinfo = (float*)(ws + WS_QINFO);
  float*    thr   = (float*)(ws + WS_THR);
  unsigned* mm    = (unsigned*)(ws + WS_MM);
  float*    Fm    = (float*)(ws + WS_F);
  float*    acc   = (float*)(ws + WS_ACC);
  float*    pose  = (float*)(ws + WS_POSE);

  init_kernel<<<129, 256, 0, stream>>>(ghist, acc, mm, pose);
  convert_kernel<<<(BATCH*NPTS*DDIM)/256, 256, 0, stream>>>(desc1, desc2, d1h, d2h);
  rowmax_kernel<<<BATCH*NPTS, 256, 0, stream>>>(matches, rmax, mm);
  hist1_kernel<<<BATCH*32, 256, 0, stream>>>(matches, mm, ghist);
  scan1_kernel<<<BATCH, 256, 0, stream>>>(ghist, mm, qinfo);
  hist2_kernel<<<BATCH*32, 256, 0, stream>>>(matches, mm, qinfo, ghist);
  scan2_kernel<<<BATCH, 32, 0, stream>>>(ghist, qinfo, thr);
  poseF_kernel<<<1, 32, 0, stream>>>(R_pred, t_pred, R_gt, t_gt, Kmat, Fm, pose);
  precompute_kernel<<<(BATCH*NPTS)/256, 256, 0, stream>>>(kp1, kp2, R_gt, Fm, p2t, fx1, d1s, d2s);
  corr_epi_kernel<<<BATCH*NPTS, 256, 0, stream>>>(matches, kp2, p2t, fx1, d1s, d2s, thr, acc);
  contrastive_kernel<<<BATCH*(NPTS/16), 32, 0, stream>>>(matches, d1h, d2h, rmax, acc);
  finalize_kernel<<<1, 32, 0, stream>>>(acc, pose, (float*)d_out);
}
